// QGNN_EM_28217935135259
// MI455X (gfx1250) — matrix operand order repaired and checked by reading, not compiled, not run on hardware
//
#include <hip/hip_runtime.h>
#include <hip/hip_bf16.h>

typedef __attribute__((ext_vector_type(16))) _Float16 v16h;
typedef __attribute__((ext_vector_type(8)))  float    v8f;
typedef __attribute__((ext_vector_type(4)))  unsigned int u32x4;
typedef __attribute__((ext_vector_type(8)))  int      i32x8;
typedef __attribute__((ext_vector_type(4)))  int      i32x4;

__device__ __forceinline__ float silu_f(float x) { return x / (1.0f + __expf(-x)); }

#define KT1 5   // ceil(134/32) K-tiles, padded to 160
#define NT1 8   // 128/16 N-tiles (MSG_DIM)
#define KT2 4   // 128/32 K-tiles for layer 2
#define TPW 8   // tiles (of 16 edges) per wave per block
#define W1FRAGS (NT1 * KT1)           // 40 fragments
#define WFRAGS  (W1FRAGS + KT2)       // 44 fragments total (W1 then W2)
#define WHALVES (WFRAGS * 32 * 16)    // 22528 f16 elements = 44 KB

// ---------------------------------------------------------------------------
// Kernel 0: zero the aggregation workspace
// ---------------------------------------------------------------------------
__global__ void qgnn_zero_kernel(float* __restrict__ p, int n) {
    int i = blockIdx.x * blockDim.x + threadIdx.x;
    if (i < n) p[i] = 0.0f;
}

// ---------------------------------------------------------------------------
// Kernel 0b: one-time weight prep: fp32 -> f16, pre-swizzled into per-lane
// WMMA B-fragment order, written to workspace (TDM source image).
// ---------------------------------------------------------------------------
__global__ void qgnn_prep_kernel(const float* __restrict__ W1,
                                 const float* __restrict__ W2,
                                 _Float16* __restrict__ wh) {
    int idx = blockIdx.x * blockDim.x + threadIdx.x;
    if (idx < 160 * 128) {                       // W1 padded 134->160
        int k = idx >> 7, n = idx & 127;
        float v = (k < 134) ? W1[k * 128 + n] : 0.0f;
        int t = k >> 5, kk = k & 31;
        int lane = ((kk & 8) << 1) | (n & 15);
        int j = (kk & 7) | ((kk & 16) >> 1);
        int nt = n >> 4;
        wh[(((nt * KT1) + t) * 32 + lane) * 16 + j] = (_Float16)v;
    } else if (idx < 160 * 128 + 128 * 16) {     // W2
        int i2 = idx - 160 * 128;
        int k = i2 >> 4, n = i2 & 15;
        float v = W2[k * 16 + n];
        int t = k >> 5, kk = k & 31;
        int lane = ((kk & 8) << 1) | n;
        int j = (kk & 7) | ((kk & 16) >> 1);
        wh[W1FRAGS * 32 * 16 + (t * 32 + lane) * 16 + j] = (_Float16)v;
    }
}

// ---------------------------------------------------------------------------
// Kernel 1: edge message MLP  cat[E,134] -> SiLU(W1) -> SiLU(W2) -> scatter-add
// 4 waves / block; each wave iterates over TPW 16-edge tiles. Weights arrive
// in LDS via one Tensor Data Mover DMA (tensor_load_to_lds, TENSORcnt).
// ---------------------------------------------------------------------------
__global__ __launch_bounds__(128) void qgnn_edge_kernel(
    const float* __restrict__ h_i, const float* __restrict__ node_params,
    const float* __restrict__ edge_params,
    const _Float16* __restrict__ wh,      // pre-swizzled f16 W1|W2 image (global)
    const float* __restrict__ b1, const float* __restrict__ b2,
    const int* __restrict__ edge_index, float* __restrict__ agg, int E)
{
    __shared__ v16h sWv[WFRAGS * 32];         // 44 KB: W1 frags then W2 frags
    __shared__ float sB1[128];
    __shared__ float sB2[16];
    __shared__ v16h sHidV[4 * 16 * 128 / 16]; // 16 KB: 4 waves x (16x128 f16)

    const int tid = threadIdx.x;

    // --- stage weights with the TDM: one 22528-element f16 tile -> LDS ---
    if (tid == 0) {
        unsigned lds_off = (unsigned)(size_t)(&sWv[0]);   // flat addr low 32b = LDS offset
        unsigned long long ga = (unsigned long long)(size_t)wh;
        const unsigned nelem = WHALVES;                   // 22528 (< 2^16)
        u32x4 g0;
        g0[0] = 1u;                                        // count=1 (valid), load
        g0[1] = lds_off;                                   // lds_addr
        g0[2] = (unsigned)(ga & 0xFFFFFFFFull);            // global_addr[31:0]
        g0[3] = (unsigned)((ga >> 32) & 0x01FFFFFFull) | 0x80000000u; // addr[56:32] | type=2
        i32x8 g1 = {};
        g1[0] = 0x00010000;                                // data_size=1 -> 2 bytes
        g1[1] = (int)((nelem & 0xFFFFu) << 16);            // tensor_dim0[15:0]
        g1[2] = (int)(((nelem >> 16) & 0xFFFFu) | (1u << 16)); // dim0 hi | tensor_dim1=1
        g1[3] = (int)((nelem & 0xFFFFu) << 16);            // tile_dim0
        g1[4] = 1;                                         // tile_dim1=1
        g1[5] = (int)nelem;                                // tensor_dim0_stride
        i32x4 gz = {};
#if defined(__clang_major__) && (__clang_major__ >= 23)
        i32x8 gz8 = {};
        __builtin_amdgcn_tensor_load_to_lds(g0, g1, gz, gz, gz8, 0);
#else
        __builtin_amdgcn_tensor_load_to_lds(g0, g1, gz, gz, 0);
#endif
        __builtin_amdgcn_s_wait_tensorcnt(0);
    }
    if (tid < 128) sB1[tid] = b1[tid];
    if (tid < 16)  sB2[tid] = b2[tid];
    __syncthreads();

    const int wave = tid >> 5;
    const int lane = tid & 31;
    const int mrow = lane & 15;                        // A-matrix row handled by lane
    const int kb   = (lane >> 4) * 8;                  // K sub-block per A layout
    const int nlo  = lane & 15;                        // D-matrix column of lane
    const int mhi  = (lane >> 4) * 8;                  // D-matrix row offset
    const int* recv_idx = edge_index + E;
    _Float16* hid = (_Float16*)sHidV + wave * (16 * 128);

    const int waveBase = (blockIdx.x * 4 + wave) * (16 * TPW);

#pragma unroll 1
    for (int tile = 0; tile < TPW; ++tile) {
        const int e0 = waveBase + tile * 16;

        // prefetch next tile's edge indices (global_prefetch_b8)
        if (tile + 1 < TPW) {
            int en = e0 + 16 + mrow;
            if (en < E) {
                __builtin_prefetch(&edge_index[en], 0, 0);
                __builtin_prefetch(&recv_idx[en], 0, 0);
            }
        }

        int e = e0 + mrow;
        int eS = (e < E) ? e : 0;
        int s = edge_index[eS];
        int r = recv_idx[eS];

        // --- build A fragments (16x32 f16 each) for the padded 134->160 input ---
        v16h a1[KT1];
        const float* hs = h_i + (size_t)s * 64;
        const float* hr = h_i + (size_t)r * 64;
#pragma unroll
        for (int t = 0; t < 4; ++t) {
            const float* src = (t < 2) ? hs : hr;
            int off = ((t & 1) ? 32 : 0) + kb;
            v16h a;
#pragma unroll
            for (int j = 0; j < 8; ++j) a[j] = (_Float16)src[off + j];
#pragma unroll
            for (int j = 0; j < 8; ++j) a[8 + j] = (_Float16)src[off + 16 + j];
            a1[t] = a;
        }
        {   // last K-tile: node_params / edge_params + zero padding (k=128..159)
            v16h a = {};
            if (kb == 0) {
                a[0] = (_Float16)node_params[s * 2 + 0];
                a[1] = (_Float16)node_params[s * 2 + 1];
                a[2] = (_Float16)node_params[r * 2 + 0];
                a[3] = (_Float16)node_params[r * 2 + 1];
                // faithful torch gather: ep[e] = edge_params[edge_index.flat[2e..2e+1]]
                a[4] = (_Float16)edge_params[edge_index[2 * eS + 0]];
                a[5] = (_Float16)edge_params[edge_index[2 * eS + 1]];
            }
            a1[4] = a;
        }

        // --- layer 1: [16x160] @ [160x128] via 40 WMMAs ---
#pragma unroll
        for (int nt = 0; nt < NT1; ++nt) {
            v8f acc = {};
#pragma unroll
            for (int t = 0; t < KT1; ++t) {
                v16h bf = sWv[(nt * KT1 + t) * 32 + lane];
                acc = __builtin_amdgcn_wmma_f32_16x16x32_f16(
                    false, a1[t], false, bf, (short)0, acc, false, false);
            }
            float bb = sB1[nt * 16 + nlo];
#pragma unroll
            for (int rr = 0; rr < 8; ++rr) {
                float x = acc[rr] + bb;
                hid[(mhi + rr) * 128 + nt * 16 + nlo] = (_Float16)silu_f(x);
            }
        }

        // --- lane-transpose through LDS: rebuild A fragments for layer 2 ---
        v16h a2[KT2];
#pragma unroll
        for (int t = 0; t < KT2; ++t) {
            const _Float16* src = hid + mrow * 128 + t * 32 + kb;
            v16h a;
#pragma unroll
            for (int j = 0; j < 8; ++j) a[j] = src[j];
#pragma unroll
            for (int j = 0; j < 8; ++j) a[8 + j] = src[16 + j];
            a2[t] = a;
        }

        // --- layer 2: [16x128] @ [128x16] via 4 WMMAs ---
        v8f acc2 = {};
#pragma unroll
        for (int t = 0; t < KT2; ++t) {
            v16h bf = sWv[(W1FRAGS + t) * 32 + lane];
            acc2 = __builtin_amdgcn_wmma_f32_16x16x32_f16(
                false, a2[t], false, bf, (short)0, acc2, false, false);
        }

        // --- SiLU + scatter-add into agg[receiver] ---
        float bb2 = sB2[nlo];
#pragma unroll
        for (int rr = 0; rr < 8; ++rr) {
            int ee = e0 + mhi + rr;
            if (ee < E) {
                float x = silu_f(acc2[rr] + bb2);
                int rv = recv_idx[ee];
                atomicAdd(&agg[(size_t)rv * 16 + nlo], x);
            }
        }
    }
}

// ---------------------------------------------------------------------------
// Kernel 2: per-node eigh (Jacobi 8x8) + WMMA node MLP + softmax + V diag(l) V^T
// ---------------------------------------------------------------------------
#define CE(i, j)                                                         \
    do {                                                                 \
        if (w[i] > w[j]) {                                               \
            float tw = w[i]; w[i] = w[j]; w[j] = tw;                     \
            for (int kz = 0; kz < 8; ++kz) {                             \
                float tv = Vv[kz][i]; Vv[kz][i] = Vv[kz][j]; Vv[kz][j] = tv; \
            }                                                            \
        }                                                                \
    } while (0)

__global__ __launch_bounds__(128) void qgnn_node_kernel(
    const float* __restrict__ h_i, const float* __restrict__ agg,
    const float* __restrict__ W3, const float* __restrict__ b3,
    const float* __restrict__ W4, const float* __restrict__ b4,
    float* __restrict__ out, int Nn)
{
    __shared__ v16h sW3v[NT1 * 32];      // 8 KB: K=32 (1 tile) x 8 N-tiles
    __shared__ v16h sW4v[KT2 * 32];      // 4 KB: K=128, N padded 8->16
    __shared__ float sB3[128];
    __shared__ float sB4[16];
    __shared__ v16h sFeatV[128 * 32 / 16];   // 8 KB : 128 nodes x 32 f16 feats
    __shared__ v16h sHid2V[128 * 128 / 16];  // 32 KB: 128 nodes x 128 f16 hidden
    __shared__ float sLam[128 * 8];          // 4 KB : raw logits

    const int tid = threadIdx.x;

    _Float16* sW3 = (_Float16*)sW3v;
    for (int idx = tid; idx < 32 * 128; idx += blockDim.x) {
        int k = idx >> 7, n = idx & 127;
        float v = W3[k * 128 + n];
        int lane = ((k & 8) << 1) | (n & 15);
        int j = (k & 7) | ((k & 16) >> 1);
        int nt = n >> 4;
        sW3[(nt * 32 + lane) * 16 + j] = (_Float16)v;
    }
    _Float16* sW4 = (_Float16*)sW4v;
    for (int idx = tid; idx < 128 * 16; idx += blockDim.x) {
        int k = idx >> 4, n = idx & 15;
        float v = (n < 8) ? W4[k * 8 + n] : 0.0f;
        int t = k >> 5, kk = k & 31;
        int lane = ((kk & 8) << 1) | n;
        int j = (kk & 7) | ((kk & 16) >> 1);
        sW4[(t * 32 + lane) * 16 + j] = (_Float16)v;
    }
    if (tid < 128) sB3[tid] = b3[tid];
    if (tid < 16)  sB4[tid] = (tid < 8) ? b4[tid] : 0.0f;

    const int v = blockIdx.x * 128 + tid;
    const bool vb = (v < Nn);
    const int vs = vb ? v : 0;

    // ---- load symmetric 8x8 and run cyclic Jacobi (registers only) ----
    float A[8][8], Vv[8][8];
    const float* src = h_i + (size_t)vs * 64;
#pragma unroll
    for (int i = 0; i < 8; ++i)
#pragma unroll
        for (int j = 0; j < 8; ++j) {
            A[i][j] = src[i * 8 + j];
            Vv[i][j] = (i == j) ? 1.0f : 0.0f;
        }

#pragma unroll 1
    for (int sw = 0; sw < 6; ++sw) {
#pragma unroll
        for (int p = 0; p < 7; ++p) {
#pragma unroll
            for (int q = p + 1; q < 8; ++q) {
                float apq = A[p][q];
                if (__builtin_fabsf(apq) > 1e-12f) {
                    float theta = 0.5f * (A[q][q] - A[p][p]) / apq;
                    float sgn = (theta >= 0.0f) ? 1.0f : -1.0f;
                    float t = sgn / (__builtin_fabsf(theta) + __builtin_sqrtf(1.0f + theta * theta));
                    float c = 1.0f / __builtin_sqrtf(1.0f + t * t);
                    float s = t * c;
#pragma unroll
                    for (int k = 0; k < 8; ++k) {     // columns p,q
                        float akp = A[k][p], akq = A[k][q];
                        A[k][p] = c * akp - s * akq;
                        A[k][q] = s * akp + c * akq;
                    }
#pragma unroll
                    for (int k = 0; k < 8; ++k) {     // rows p,q
                        float apk = A[p][k], aqk = A[q][k];
                        A[p][k] = c * apk - s * aqk;
                        A[q][k] = s * apk + c * aqk;
                    }
#pragma unroll
                    for (int k = 0; k < 8; ++k) {     // eigenvector columns
                        float vkp = Vv[k][p], vkq = Vv[k][q];
                        Vv[k][p] = c * vkp - s * vkq;
                        Vv[k][q] = s * vkp + c * vkq;
                    }
                }
            }
        }
    }
    float w[8];
#pragma unroll
    for (int j = 0; j < 8; ++j) w[j] = A[j][j];

    // ---- ascending sort (Batcher network, compile-time indices) ----
    CE(0,1); CE(2,3); CE(4,5); CE(6,7);
    CE(0,2); CE(1,3); CE(4,6); CE(5,7);
    CE(1,2); CE(5,6);
    CE(0,4); CE(1,5); CE(2,6); CE(3,7);
    CE(2,4); CE(3,5);
    CE(1,2); CE(3,4); CE(5,6);

    // ---- feats = [agg(16), eigvals(8), rownorm(eigvecs)(8)] -> LDS f16 ----
    _Float16* sFeat = (_Float16*)sFeatV;
    {
        _Float16* f = sFeat + tid * 32;
        const float* ag = agg + (size_t)vs * 16;
#pragma unroll
        for (int j = 0; j < 16; ++j) f[j] = (_Float16)(vb ? ag[j] : 0.0f);
#pragma unroll
        for (int j = 0; j < 8; ++j) f[16 + j] = (_Float16)w[j];
#pragma unroll
        for (int j = 0; j < 8; ++j) {
            float m2 = 0.0f;
#pragma unroll
            for (int k = 0; k < 8; ++k) m2 += Vv[j][k] * Vv[j][k];
            f[24 + j] = (_Float16)__builtin_sqrtf(m2);
        }
    }
    __syncthreads();

    // ---- WMMA node MLP: each wave handles two 16-node tiles ----
    const int lane = tid & 31;
    const int wave = tid >> 5;
    const int mrow = lane & 15;
    const int kb   = (lane >> 4) * 8;
    const int nlo  = lane & 15;
    const int mhi  = (lane >> 4) * 8;
    _Float16* sHid2 = (_Float16*)sHid2V;

#pragma unroll 1
    for (int tt = 0; tt < 2; ++tt) {
        int rowBase = (wave * 2 + tt) * 16;
        // A fragment (16 nodes x 32 feats, exactly one K-tile)
        v16h af;
        {
            const _Float16* fr = sFeat + (rowBase + mrow) * 32 + kb;
#pragma unroll
            for (int j = 0; j < 8; ++j) af[j] = fr[j];
#pragma unroll
            for (int j = 0; j < 8; ++j) af[8 + j] = fr[16 + j];
        }
        // layer 3: [16x32] @ [32x128]
#pragma unroll
        for (int nt = 0; nt < NT1; ++nt) {
            v8f acc = {};
            acc = __builtin_amdgcn_wmma_f32_16x16x32_f16(
                false, af, false, sW3v[nt * 32 + lane], (short)0, acc, false, false);
            float bb = sB3[nt * 16 + nlo];
#pragma unroll
            for (int rr = 0; rr < 8; ++rr) {
                float x = acc[rr] + bb;
                sHid2[(rowBase + mhi + rr) * 128 + nt * 16 + nlo] = (_Float16)silu_f(x);
            }
        }
        // layer 4: [16x128] @ [128x16(pad)]
        v8f acc2 = {};
#pragma unroll
        for (int t = 0; t < KT2; ++t) {
            const _Float16* hr = sHid2 + (rowBase + mrow) * 128 + t * 32 + kb;
            v16h a2;
#pragma unroll
            for (int j = 0; j < 8; ++j) a2[j] = hr[j];
#pragma unroll
            for (int j = 0; j < 8; ++j) a2[8 + j] = hr[16 + j];
            acc2 = __builtin_amdgcn_wmma_f32_16x16x32_f16(
                false, a2, false, sW4v[t * 32 + lane], (short)0, acc2, false, false);
        }
        if (nlo < 8) {
            float bb = sB4[nlo];
#pragma unroll
            for (int rr = 0; rr < 8; ++rr)
                sLam[(rowBase + mhi + rr) * 8 + nlo] = acc2[rr] + bb;
        }
    }
    __syncthreads();

    // ---- softmax + reconstruction out = V diag(lam) V^T ----
    float lam[8];
    {
        float mx = -3.0e38f;
#pragma unroll
        for (int j = 0; j < 8; ++j) { float l = sLam[tid * 8 + j]; lam[j] = l; mx = fmaxf(mx, l); }
        float sum = 0.0f;
#pragma unroll
        for (int j = 0; j < 8; ++j) { lam[j] = __expf(lam[j] - mx); sum += lam[j]; }
        float inv = 1.0f / sum;
#pragma unroll
        for (int j = 0; j < 8; ++j) lam[j] *= inv;
    }
    if (vb) {
        float Wl[8][8];
#pragma unroll
        for (int i = 0; i < 8; ++i)
#pragma unroll
            for (int j = 0; j < 8; ++j) Wl[i][j] = Vv[i][j] * lam[j];
        float* o = out + (size_t)v * 64;
#pragma unroll
        for (int i = 0; i < 8; ++i)
#pragma unroll
            for (int k = 0; k < 8; ++k) {
                float sacc = 0.0f;
#pragma unroll
                for (int j = 0; j < 8; ++j) sacc += Wl[i][j] * Vv[k][j];
                o[i * 8 + k] = sacc;
            }
    }
}

// ---------------------------------------------------------------------------
extern "C" void kernel_launch(void* const* d_in, const int* in_sizes, int n_in,
                              void* d_out, int out_size, void* d_ws, size_t ws_size,
                              hipStream_t stream) {
    const float* h_i         = (const float*)d_in[0];
    const float* node_params = (const float*)d_in[1];
    const float* edge_params = (const float*)d_in[2];
    const float* W1 = (const float*)d_in[3];
    const float* b1 = (const float*)d_in[4];
    const float* W2 = (const float*)d_in[5];
    const float* b2 = (const float*)d_in[6];
    const float* W3 = (const float*)d_in[7];
    const float* b3 = (const float*)d_in[8];
    const float* W4 = (const float*)d_in[9];
    const float* b4 = (const float*)d_in[10];
    const int* edge_index = (const int*)d_in[11];

    int Nn = in_sizes[0] / 64;   // h_i: [N,8,8]
    int E  = in_sizes[11] / 2;   // edge_index: [2,E]

    // workspace layout: agg [N*16 f32] | pre-swizzled f16 weights (44 KB)
    float* agg = (float*)d_ws;
    size_t aggBytes = (((size_t)Nn * 16 * sizeof(float)) + 255) & ~(size_t)255;
    _Float16* wh = (_Float16*)((char*)d_ws + aggBytes);
    float* out = (float*)d_out;

    int aggN = Nn * 16;
    qgnn_zero_kernel<<<(aggN + 255) / 256, 256, 0, stream>>>(agg, aggN);

    int prepN = 160 * 128 + 128 * 16;
    qgnn_prep_kernel<<<(prepN + 255) / 256, 256, 0, stream>>>(W1, W2, wh);

    // 512 edges per 128-thread block (4 waves x 8 tiles x 16 edges)
    int eblocks = (E + (64 * TPW) - 1) / (64 * TPW);
    qgnn_edge_kernel<<<eblocks, 128, 0, stream>>>(
        h_i, node_params, edge_params, wh, b1, b2, edge_index, agg, E);

    int nblocks = (Nn + 127) / 128;
    qgnn_node_kernel<<<nblocks, 128, 0, stream>>>(
        h_i, agg, W3, b3, W4, b4, out, Nn);
}
